// AutoCorrelation_12060268168024
// MI455X (gfx1250) — matrix operand order repaired and checked by reading, not compiled, not run on hardware
//
#include <hip/hip_runtime.h>

typedef __attribute__((ext_vector_type(16))) _Float16 v16h;
typedef __attribute__((ext_vector_type(8)))  float    v8f;

#define B_    4
#define L_    2048
#define D_    512
#define BL_   8192        // B_*L_
#define HL_   16384       // H*Lq = 8*2048 (mean divisor)

#if defined(__has_builtin)
# if __has_builtin(__builtin_amdgcn_global_load_async_to_lds_b128) && \
     __has_builtin(__builtin_amdgcn_s_wait_asynccnt)
#  define AC_ASYNC_LDS 1
# endif
#endif
#ifndef AC_ASYNC_LDS
# define AC_ASYNC_LDS 0
#endif

#if AC_ASYNC_LDS
typedef int ac_v4i __attribute__((ext_vector_type(4)));
typedef __attribute__((address_space(1))) ac_v4i* ac_gptr;   // global (AS1) int4*
typedef __attribute__((address_space(3))) ac_v4i* ac_lptr;   // LDS (AS3) int4*
#endif

// ---------------------------------------------------------------------------
// 1) qsum partial: part[b,chunk,c] = sum over 256 rows of queries[b, :, c]
// ---------------------------------------------------------------------------
__global__ void ac_qsum_partial(const float* __restrict__ q, float* __restrict__ part) {
    int c     = threadIdx.x;      // 0..511
    int chunk = blockIdx.x;       // 0..7
    int b     = blockIdx.y;       // 0..3
    const float* p = q + ((size_t)b * L_ + (size_t)chunk * 256) * D_ + c;
    float s = 0.f;
    #pragma unroll 8
    for (int i = 0; i < 256; ++i) s += p[(size_t)i * D_];
    part[((size_t)b * 8 + chunk) * D_ + c] = s;
}

__global__ void ac_qsum_reduce(const float* __restrict__ part, float* __restrict__ qs) {
    int b = blockIdx.x, c = threadIdx.x;
    float s = 0.f;
    #pragma unroll
    for (int ch = 0; ch < 8; ++ch) s += part[((size_t)b * 8 + ch) * D_ + c];
    qs[(size_t)b * D_ + c] = s;
}

// u[b,c] = sum_d qs[b,d] * wq[d,c]
__global__ void ac_proj_u(const float* __restrict__ qs, const float* __restrict__ wq,
                          float* __restrict__ u) {
    int b = blockIdx.x, c = threadIdx.x;
    float s = 0.f;
    for (int d = 0; d < D_; ++d) s += qs[(size_t)b * D_ + d] * wq[(size_t)d * D_ + c];
    u[(size_t)b * D_ + c] = s;
}

// t[b,d] = sum_c wk[d,c] * u[b,c]
__global__ void ac_proj_t(const float* __restrict__ u, const float* __restrict__ wk,
                          float* __restrict__ t) {
    int b = blockIdx.x, d = threadIdx.x;
    float s = 0.f;
    for (int c = 0; c < D_; ++c) s += wk[(size_t)d * D_ + c] * u[(size_t)b * D_ + c];
    t[(size_t)b * D_ + d] = s;
}

// ---------------------------------------------------------------------------
// 2) scores[b,j] = (1/16384) * dot(keys[b,j,:], t[b,:])   (one wave per row)
// ---------------------------------------------------------------------------
__global__ void ac_scores(const float* __restrict__ keys, const float* __restrict__ t,
                          float* __restrict__ scores) {
    int gw   = (int)((blockIdx.x * blockDim.x + threadIdx.x) >> 5);  // row over B*L
    int lane = threadIdx.x & 31;
    int b    = gw >> 11;
    const float* krow = keys + (size_t)gw * D_;
    const float* tb   = t + (size_t)b * D_;
    float s = 0.f;
    #pragma unroll
    for (int i = lane; i < D_; i += 32) s += krow[i] * tb[i];
    #pragma unroll
    for (int off = 16; off > 0; off >>= 1) s += __shfl_down(s, off, 32);
    if (lane == 0) scores[gw] = s * (1.0f / (float)HL_);
}

// ---------------------------------------------------------------------------
// 3) top-4 + softmax per batch (tie-break: lower index wins, like lax.top_k)
// ---------------------------------------------------------------------------
__global__ void ac_topk4(const float* __restrict__ scores, int* __restrict__ outIdx,
                         float* __restrict__ outW) {
    __shared__ float sv[L_];
    __shared__ float rv[256];
    __shared__ int   ri[256];
    int b = blockIdx.x, tid = threadIdx.x;
    for (int i = tid; i < L_; i += 256) sv[i] = scores[(size_t)b * L_ + i];
    __syncthreads();
    float vals[4]; int idxs[4];
    for (int kk = 0; kk < 4; ++kk) {
        float best = -__builtin_inff(); int bi = 0x7fffffff;
        for (int i = tid; i < L_; i += 256) {
            float v = sv[i];
            if (v > best || (v == best && i < bi)) { best = v; bi = i; }
        }
        rv[tid] = best; ri[tid] = bi;
        __syncthreads();
        for (int s = 128; s > 0; s >>= 1) {
            if (tid < s) {
                if (rv[tid + s] > rv[tid] ||
                    (rv[tid + s] == rv[tid] && ri[tid + s] < ri[tid])) {
                    rv[tid] = rv[tid + s]; ri[tid] = ri[tid + s];
                }
            }
            __syncthreads();
        }
        vals[kk] = rv[0]; idxs[kk] = ri[0];
        __syncthreads();
        if (tid == 0) sv[idxs[kk]] = -__builtin_inff();
        __syncthreads();
    }
    if (tid == 0) {
        float mx = vals[0];           // descending, so vals[0] is the max
        float e[4], s = 0.f;
        #pragma unroll
        for (int kk = 0; kk < 4; ++kk) { e[kk] = expf(vals[kk] - mx); s += e[kk]; }
        #pragma unroll
        for (int kk = 0; kk < 4; ++kk) {
            outIdx[b * 4 + kk] = idxs[kk];
            outW[b * 4 + kk]   = e[kk] / s;
        }
    }
}

// ---------------------------------------------------------------------------
// 4a) f32 -> f16 convert (plain)
// ---------------------------------------------------------------------------
__global__ void ac_cvt_h(const float* __restrict__ in, _Float16* __restrict__ out, int n) {
    int i = (int)(blockIdx.x * 256 + threadIdx.x);
    if (i < n) out[i] = (_Float16)in[i];
}

// ---------------------------------------------------------------------------
// 4b) f32 -> f16 transpose: Bt[n][k] = (half) W[k][n]   (512x512, LDS-tiled)
// ---------------------------------------------------------------------------
__global__ void ac_cvt_h_t(const float* __restrict__ W, _Float16* __restrict__ Bt) {
    __shared__ float tile[32][33];
    int tx = threadIdx.x & 31, ty = threadIdx.x >> 5;   // 256 threads: ty 0..7
    int k0 = blockIdx.y * 32, n0 = blockIdx.x * 32;
    #pragma unroll
    for (int i = 0; i < 32; i += 8)
        tile[ty + i][tx] = W[(size_t)(k0 + ty + i) * D_ + n0 + tx];
    __syncthreads();
    #pragma unroll
    for (int i = 0; i < 32; i += 8)
        Bt[(size_t)(n0 + ty + i) * D_ + k0 + tx] = (_Float16)tile[tx][ty + i];
}

// ---------------------------------------------------------------------------
// 5) WMMA GEMM: C[M,N] f32 = A[M,K] f16 * Bt[N,K] f16  (B pre-transposed)
//    block = 256 threads (8 waves), block tile 128x64, K-slab 32.
//    Double-buffered LDS (one barrier per slab); slabs staged either by
//    GLOBAL_LOAD_ASYNC_TO_LDS_B128 (ASYNCcnt) or a register pipeline.
// ---------------------------------------------------------------------------
__global__ __launch_bounds__(256) void ac_wmma_gemm(
        const _Float16* __restrict__ A, const _Float16* __restrict__ Bt,
        float* __restrict__ C, int N, int K) {
    __shared__ _Float16 As[2][128 * 32];   // [m][k]
    __shared__ _Float16 Bs[2][64 * 32];    // [n][k]   <- transposed slab

    const int tid      = threadIdx.x;
    const int lane     = tid & 31;
    const int wvid     = tid >> 5;
    const int blockRow = blockIdx.y * 128;
    const int colBase  = blockIdx.x * 64;
    const int nslabs   = K >> 5;

    // global-load coordinates (per thread)
    const int arow0 = (tid * 2) >> 2,     aq0 = (tid * 2) & 3;       // A elem 0
    const int arow1 = (tid * 2 + 1) >> 2, aq1 = (tid * 2 + 1) & 3;   // A elem 1
    const int brow  = tid >> 2,           bq  = tid & 3;             // Bt elem

    v8f acc[4] = {};

    const int m0 = wvid * 16;
    const int m  = lane & 15;
    const int kh = (lane & 16) >> 1;   // A: lanes 16..31 take k+8   (ISA A 16x32 layout)
    const int ko = (lane & 16) >> 1;   // B: lanes 16..31 take k+8   (same k order as A)

    struct HPair { uint4 lo, hi; };

    auto compute = [&](int buf) {
        // A operand: halves [m][kh..kh+7] and [m][16+kh..23+kh] -> 2x ds_load_b128
        HPair pa;
        pa.lo = *reinterpret_cast<const uint4*>(&As[buf][(m0 + m) * 32 + kh]);
        pa.hi = *reinterpret_cast<const uint4*>(&As[buf][(m0 + m) * 32 + 16 + kh]);
        v16h a = __builtin_bit_cast(v16h, pa);
        #pragma unroll
        for (int nt = 0; nt < 4; ++nt) {
            int nn = (nt << 4) | (lane & 15);
            HPair pb;
            pb.lo = *reinterpret_cast<const uint4*>(&Bs[buf][nn * 32 + ko]);
            pb.hi = *reinterpret_cast<const uint4*>(&Bs[buf][nn * 32 + ko + 16]);
            v16h bm = __builtin_bit_cast(v16h, pb);
            acc[nt] = __builtin_amdgcn_wmma_f32_16x16x32_f16(
                false, a, false, bm, (short)0, acc[nt], false, false);
        }
    };

#if AC_ASYNC_LDS
    auto loadAsync = [&](int kc, int buf) {
        __builtin_amdgcn_global_load_async_to_lds_b128(
            (ac_gptr)(A + (size_t)(blockRow + arow0) * K + kc * 32 + aq0 * 8),
            (ac_lptr)&As[buf][arow0 * 32 + aq0 * 8], 0, 0);
        __builtin_amdgcn_global_load_async_to_lds_b128(
            (ac_gptr)(A + (size_t)(blockRow + arow1) * K + kc * 32 + aq1 * 8),
            (ac_lptr)&As[buf][arow1 * 32 + aq1 * 8], 0, 0);
        __builtin_amdgcn_global_load_async_to_lds_b128(
            (ac_gptr)(Bt + (size_t)(colBase + brow) * K + kc * 32 + bq * 8),
            (ac_lptr)&Bs[buf][brow * 32 + bq * 8], 0, 0);
    };

    loadAsync(0, 0);
    __builtin_amdgcn_s_wait_asynccnt(0);
    __syncthreads();
    for (int kc = 0; kc < nslabs; ++kc) {
        const int  buf  = kc & 1;
        const bool more = (kc + 1 < nslabs);
        if (more) loadAsync(kc + 1, buf ^ 1);   // async engine fills idle buffer
        compute(buf);
        if (more) __builtin_amdgcn_s_wait_asynccnt(0);
        __syncthreads();
    }
#else
    uint4 aReg0, aReg1, bReg;
    auto loadG = [&](int kc) {
        aReg0 = *reinterpret_cast<const uint4*>(
            A + (size_t)(blockRow + arow0) * K + kc * 32 + aq0 * 8);
        aReg1 = *reinterpret_cast<const uint4*>(
            A + (size_t)(blockRow + arow1) * K + kc * 32 + aq1 * 8);
        bReg  = *reinterpret_cast<const uint4*>(
            Bt + (size_t)(colBase + brow) * K + kc * 32 + bq * 8);
    };
    auto storeL = [&](int buf) {
        *reinterpret_cast<uint4*>(&As[buf][arow0 * 32 + aq0 * 8]) = aReg0;
        *reinterpret_cast<uint4*>(&As[buf][arow1 * 32 + aq1 * 8]) = aReg1;
        *reinterpret_cast<uint4*>(&Bs[buf][brow * 32 + bq * 8])   = bReg;
    };

    loadG(0);
    storeL(0);
    __syncthreads();
    for (int kc = 0; kc < nslabs; ++kc) {
        const int  buf  = kc & 1;
        const bool more = (kc + 1 < nslabs);
        if (more) loadG(kc + 1);
        compute(buf);
        if (more) storeL(buf ^ 1);   // write idle buffer; reads of it done pre-barrier
        __syncthreads();
    }
#endif

    // store C (ISA 32-bit C/D 16x16 layout: VGPR r -> M=r (+8 for lanes 16..31))
    const int rbase = (lane & 16) >> 1;
    #pragma unroll
    for (int nt = 0; nt < 4; ++nt) {
        #pragma unroll
        for (int r = 0; r < 8; ++r) {
            C[(size_t)(blockRow + m0 + rbase + r) * N + colBase + nt * 16 + (lane & 15)]
                = acc[nt][r];
        }
    }
}

// ---------------------------------------------------------------------------
// 6) rolled gather + transpose(0,3,1,2).reshape quirk -> M (f16) for GEMM2
//    M[b, l', c'] with f = l'*512+c': d=f>>14, h=(f>>11)&7, l=f&2047, c=h*64+d
// ---------------------------------------------------------------------------
__global__ void ac_gather(const float* __restrict__ Vproj, const int* __restrict__ tIdx,
                          const float* __restrict__ tW, _Float16* __restrict__ Mh) {
    int gid = (int)(blockIdx.x * 256 + threadIdx.x);   // 0 .. 4*1048576-1
    int b = gid >> 20;
    int f = gid & 1048575;
    int d = f >> 14;
    int h = (f >> 11) & 7;
    int l = f & 2047;
    int c = (h << 6) | d;
    const float* base = Vproj + ((size_t)b << 20);
    float s = 0.f;
    #pragma unroll
    for (int kk = 0; kk < 4; ++kk) {
        int shift = tIdx[b * 4 + kk];
        s += tW[b * 4 + kk] * base[((size_t)((l + shift) & 2047) << 9) + c];
    }
    Mh[gid] = (_Float16)s;
}

// ---------------------------------------------------------------------------
extern "C" void kernel_launch(void* const* d_in, const int* in_sizes, int n_in,
                              void* d_out, int out_size, void* d_ws, size_t ws_size,
                              hipStream_t stream) {
    const float* queries = (const float*)d_in[0];
    const float* keys    = (const float*)d_in[1];
    const float* values  = (const float*)d_in[2];
    const float* wq      = (const float*)d_in[3];
    const float* wk      = (const float*)d_in[4];
    const float* wv      = (const float*)d_in[5];
    const float* wo      = (const float*)d_in[6];
    float*       out     = (float*)d_out;

    char* ws = (char*)d_ws;
    size_t off = 0;
    auto alloc = [&](size_t bytes) -> void* {
        void* p = ws + off;
        off = (off + bytes + 255) & ~(size_t)255;
        return p;
    };
    float*    Vproj    = (float*)   alloc((size_t)BL_ * D_ * 4);   // 16.8 MB
    _Float16* values_h = (_Float16*)alloc((size_t)BL_ * D_ * 2);   //  8.4 MB
    _Float16* Mh       = (_Float16*)alloc((size_t)BL_ * D_ * 2);   //  8.4 MB
    _Float16* wv_t     = (_Float16*)alloc((size_t)D_ * D_ * 2);    // transposed f16
    _Float16* wo_t     = (_Float16*)alloc((size_t)D_ * D_ * 2);    // transposed f16
    float*    part     = (float*)   alloc((size_t)B_ * 8 * D_ * 4);
    float*    qs       = (float*)   alloc((size_t)B_ * D_ * 4);
    float*    u        = (float*)   alloc((size_t)B_ * D_ * 4);
    float*    t        = (float*)   alloc((size_t)B_ * D_ * 4);
    float*    scores   = (float*)   alloc((size_t)B_ * L_ * 4);
    int*      topIdx   = (int*)     alloc((size_t)B_ * 4 * 4);
    float*    topW     = (float*)   alloc((size_t)B_ * 4 * 4);

    // mean-corr path (algebraically collapsed)
    ac_qsum_partial<<<dim3(8, B_), 512, 0, stream>>>(queries, part);
    ac_qsum_reduce <<<B_, 512, 0, stream>>>(part, qs);
    ac_proj_u      <<<B_, 512, 0, stream>>>(qs, wq, u);
    ac_proj_t      <<<B_, 512, 0, stream>>>(u, wk, t);
    ac_scores      <<<(BL_ * 32) / 256, 256, 0, stream>>>(keys, t, scores);
    ac_topk4       <<<B_, 256, 0, stream>>>(scores, topIdx, topW);

    // f16 staging (weights transposed to [N][K] for k-contiguous LDS slabs)
    ac_cvt_h  <<<(BL_ * D_) / 256, 256, 0, stream>>>(values, values_h, BL_ * D_);
    ac_cvt_h_t<<<dim3(16, 16), 256, 0, stream>>>(wv, wv_t);
    ac_cvt_h_t<<<dim3(16, 16), 256, 0, stream>>>(wo, wo_t);

    // Vproj = values @ wv  (WMMA)
    ac_wmma_gemm<<<dim3(D_ / 64, BL_ / 128), 256, 0, stream>>>(values_h, wv_t, Vproj, D_, D_);
    // rolled weighted gather + reshape quirk
    ac_gather<<<(BL_ * D_) / 256, 256, 0, stream>>>(Vproj, topIdx, topW, Mh);
    // out = M @ wo  (WMMA)
    ac_wmma_gemm<<<dim3(D_ / 64, BL_ / 128), 256, 0, stream>>>(Mh, wo_t, out, D_, D_);
}
